// Sender_3934190043231
// MI455X (gfx1250) — matrix operand order repaired and checked by reading, not compiled, not run on hardware
//
#include <hip/hip_runtime.h>
#include <hip/hip_bf16.h>
#include <math.h>

typedef __attribute__((ext_vector_type(16))) __bf16 v16bf;
typedef __attribute__((ext_vector_type(8)))  __bf16 v8bf;
typedef __attribute__((ext_vector_type(8)))  float  v8f;
typedef __attribute__((ext_vector_type(4))) unsigned int u32x4;
typedef __attribute__((ext_vector_type(8))) int          i32x8;
typedef __attribute__((ext_vector_type(4))) int          i32x4;

#define B_  4096
#define F_  512
#define H_  256
#define E_  128
#define V_  32000
#define T_  16
#define OC_ 17        // output columns (MAX_LEN + 1 zero pad column)
#define VSPLIT_ 8     // V-dimension split for the fused logits kernel
#define NTILES_ (V_ / 16)            // 2000 column tiles
#define TPS_ (NTILES_ / VSPLIT_)     // 250 tiles per split
#define ROWPITCH_ 528                // 512B tile row + 16B TDM pad (bank stagger)
#define BUFBYTES_ (16 * ROWPITCH_)   // 8448B per staging buffer

// ---------------- WMMA fragment helpers (contiguous k order: element s of lane half h is k = 16h + s, wave32) ----------------

// A-matrix 16x32 bf16 tile from row-major [.., lda]:
// lanes 0-15 hold M=lane, K = k0..k0+15; lanes 16-31 hold K = k0+16..k0+31.
static __device__ __forceinline__ v16bf load_a_frag(const __bf16* A, int lda, int r0, int k0, int lane) {
  const int lh = (lane >> 4) & 1;
  const __bf16* p = A + (size_t)(r0 + (lane & 15)) * lda + k0 + lh * 16;
  v8bf lo = *(const v8bf*)(p);
  v8bf hi = *(const v8bf*)(p + 8);
  v16bf f;
#pragma unroll
  for (int i = 0; i < 8; ++i) { f[i] = lo[i]; f[i + 8] = hi[i]; }
  return f;
}

// B-matrix 32x16 bf16 tile. Source is W stored row-major [N, K] (we compute X @ W^T),
// so B[k][n] = W[n][k]: lane (n = lane&15) reads 16 contiguous bf16 at K = k0 + half*16.
static __device__ __forceinline__ v16bf load_b_frag(const __bf16* W, int ldw, int n0, int k0, int lane) {
  const int lh = (lane >> 4) & 1;
  const __bf16* p = W + (size_t)(n0 + (lane & 15)) * ldw + k0 + lh * 16;
  v8bf lo = *(const v8bf*)(p);
  v8bf hi = *(const v8bf*)(p + 8);
  v16bf f;
#pragma unroll
  for (int i = 0; i < 8; ++i) { f[i] = lo[i]; f[i + 8] = hi[i]; }
  return f;
}

// Same B fragment but from an LDS-staged tile with ROWPITCH_ byte row pitch.
static __device__ __forceinline__ v16bf load_b_frag_lds(const char* buf, int k0, int lane) {
  const int lh = (lane >> 4) & 1;
  const __bf16* p = (const __bf16*)(buf + (lane & 15) * ROWPITCH_ + (size_t)(k0 + lh * 16) * 2);
  v8bf lo = *(const v8bf*)(p);
  v8bf hi = *(const v8bf*)(p + 8);
  v16bf f;
#pragma unroll
  for (int i = 0; i < 8; ++i) { f[i] = lo[i]; f[i + 8] = hi[i]; }
  return f;
}

static __device__ __forceinline__ v8f wmma_bf16(v16bf a, v16bf b, v8f c) {
  return __builtin_amdgcn_wmma_f32_16x16x32_bf16(false, a, false, b, (short)0, c, false, false);
}

// ---------------- Tensor Data Mover: stage one 16x256 bf16 W_out tile into LDS ----------------
// D# per ISA 8.3/8.4: 2D tensor, data_size=2B, tensor_dim0=256 (K), dim0_stride=256,
// tensor_dim1=32000 (rows remaining), tile 256x16. pad_enable: +4 dwords every 128 dwords
// (one 512B row) -> 528B LDS row pitch, 16B aligned and bank-staggered for ds_load_b128.
static __device__ __forceinline__ void tdm_load_wout_tile(unsigned lds_addr, const __bf16* gtile) {
  unsigned long long ga = (unsigned long long)(size_t)gtile;
  u32x4 g0;
  g0[0] = 1u;                                       // count=1 valid descriptor, user mode
  g0[1] = lds_addr;                                 // LDS byte address
  g0[2] = (unsigned)(ga & 0xFFFFFFFFu);             // global_addr[31:0]
  g0[3] = (unsigned)((ga >> 32) & 0x01FFFFFFu)      // global_addr[56:32]
          | (2u << 30);                             // type=2 ("image")
  i32x8 g1;
  g1[0] = (int)((1u << 16)          // data_size = 2 bytes
              | (1u << 20)          // pad_enable
              | (6u << 22)          // pad_interval: every 128 dwords (512B row)
              | (3u << 25));        // pad_amount: 4 dwords (16B)
  g1[1] = (int)(((unsigned)H_ & 0xFFFFu) << 16);                  // tensor_dim0[15:0] @ bits 48
  g1[2] = (int)(((unsigned)H_ >> 16) | (((unsigned)V_ & 0xFFFFu) << 16)); // td0 hi | td1 lo
  g1[3] = (int)(((unsigned)V_ >> 16) | ((unsigned)H_ << 16));     // td1 hi | tile_dim0 = 256
  g1[4] = 16;                                                     // tile_dim1 = 16 rows
  g1[5] = H_;                                                     // tensor_dim0_stride = 256
  g1[6] = 0;
  g1[7] = 0;
  i32x4 z4 = {0, 0, 0, 0};
#if __clang_major__ >= 23
  i32x8 z8 = {0, 0, 0, 0, 0, 0, 0, 0};
  __builtin_amdgcn_tensor_load_to_lds(g0, g1, z4, z4, z8, 0);
#else
  __builtin_amdgcn_tensor_load_to_lds(g0, g1, z4, z4, 0);
#endif
}

// hash -> Gumbel noise for categorical sampling via Gumbel-max
static __device__ __forceinline__ float gumbel_noise(unsigned step, unsigned row, unsigned col) {
  unsigned x = step * 0x9E3779B9u ^ (row * 0x85EBCA6Bu + 0x1234567u) ^ (col * 0xC2B2AE35u);
  x ^= x >> 16; x *= 0x7FEB352Du; x ^= x >> 15; x *= 0x846CA68Bu; x ^= x >> 16;
  float u = (float)(x >> 8) * (1.0f / 16777216.0f) + 1.0e-12f;
  return -__logf(-__logf(u));
}

// ---------------- elementwise kernels ----------------

__global__ void k_f32_to_bf16(const float* __restrict__ in, __bf16* __restrict__ out, int n) {
  int i = blockIdx.x * blockDim.x + threadIdx.x;
  if (i < n) out[i] = (__bf16)in[i];
}

__global__ void k_zero_f32(float* __restrict__ p, int n) {
  int i = blockIdx.x * blockDim.x + threadIdx.x;
  if (i < n) p[i] = 0.0f;
}

__global__ void k_sos_bcast(const float* __restrict__ sos, __bf16* __restrict__ inp) {
  int i = blockIdx.x * blockDim.x + threadIdx.x;
  if (i < B_ * E_) inp[i] = (__bf16)sos[i % E_];
}

__global__ void k_embed(const __bf16* __restrict__ embb, const int* __restrict__ sample,
                        __bf16* __restrict__ inpb) {
  int i = blockIdx.x * blockDim.x + threadIdx.x;
  if (i >= B_ * E_) return;
  int r = i / E_, e = i - r * E_;
  inpb[i] = embb[(size_t)sample[r] * E_ + e];
}

// ---------------- h0 = x @ W_in^T + b_in  (bf16 in, bf16 out) ----------------
__global__ void k_h0_gemm(const __bf16* __restrict__ xb, const __bf16* __restrict__ winb,
                          const float* __restrict__ b_in, __bf16* __restrict__ hb) {
  const int lane = threadIdx.x & 31;
  const int wave = threadIdx.x >> 5;
  const int r0 = blockIdx.x * 16;
  const int n0 = (blockIdx.y * 4 + wave) * 16;
  v8f acc = {};
#pragma unroll 4
  for (int k0 = 0; k0 < F_; k0 += 32) {
    v16bf a = load_a_frag(xb, F_, r0, k0, lane);
    v16bf b = load_b_frag(winb, F_, n0, k0, lane);
    acc = wmma_bf16(a, b, acc);
  }
  const int lh = (lane >> 4) & 1;
  const int col = n0 + (lane & 15);
  const float bias = b_in[col];
#pragma unroll
  for (int j = 0; j < 8; ++j) {
    int row = r0 + lh * 8 + j;
    hb[(size_t)row * H_ + col] = (__bf16)(acc[j] + bias);
  }
}

// ---------------- gates = inp @ W_ih^T + h @ W_hh^T + b_ih + b_hh ----------------
__global__ void k_gates_gemm(const __bf16* __restrict__ inpb, const __bf16* __restrict__ wihb,
                             const __bf16* __restrict__ hb,   const __bf16* __restrict__ whhb,
                             const float* __restrict__ b_ih,  const float* __restrict__ b_hh,
                             float* __restrict__ gates) {
  const int lane = threadIdx.x & 31;
  const int wave = threadIdx.x >> 5;
  const int r0 = blockIdx.x * 16;
  const int n0 = (blockIdx.y * 4 + wave) * 16;
  v8f acc = {};
#pragma unroll
  for (int k0 = 0; k0 < E_; k0 += 32) {
    v16bf a = load_a_frag(inpb, E_, r0, k0, lane);
    v16bf b = load_b_frag(wihb, E_, n0, k0, lane);
    acc = wmma_bf16(a, b, acc);
  }
#pragma unroll
  for (int k0 = 0; k0 < H_; k0 += 32) {
    v16bf a = load_a_frag(hb, H_, r0, k0, lane);
    v16bf b = load_b_frag(whhb, H_, n0, k0, lane);
    acc = wmma_bf16(a, b, acc);
  }
  const int lh = (lane >> 4) & 1;
  const int col = n0 + (lane & 15);
  const float bias = b_ih[col] + b_hh[col];
#pragma unroll
  for (int j = 0; j < 8; ++j) {
    int row = r0 + lh * 8 + j;
    gates[(size_t)row * (4 * H_) + col] = acc[j] + bias;
  }
}

// ---------------- LSTM pointwise: i,f,g,o -> c,h ----------------
__global__ void k_lstm_pointwise(const float* __restrict__ gates, float* __restrict__ c,
                                 __bf16* __restrict__ hb) {
  int i = blockIdx.x * blockDim.x + threadIdx.x;
  if (i >= B_ * H_) return;
  int r = i / H_, n = i - r * H_;
  const float* g = gates + (size_t)r * (4 * H_);
  float ig = 1.0f / (1.0f + __expf(-g[n]));
  float fg = 1.0f / (1.0f + __expf(-g[H_ + n]));
  float gg = tanhf(g[2 * H_ + n]);
  float og = 1.0f / (1.0f + __expf(-g[3 * H_ + n]));
  float cn = fg * c[i] + ig * gg;
  c[i] = cn;
  hb[i] = (__bf16)(og * tanhf(cn));
}

// ---------------- fused logits GEMM + streaming log-softmax/Gumbel-max (partial) ----------------
// grid(32, VSPLIT_), block 256 (8 waves). Block covers 128 rows; wave w owns rows
// blockIdx.x*128 + w*16 and keeps its full A fragment (16x256 of h) in VGPRs.
// All 8 waves share each 16x256 W_out tile: wave 0 stages it with the TDM
// (tensor_load_to_lds, double buffered, s_wait_tensorcnt) and everyone reads B
// fragments back with ds_load_b128. Streaming stats (m, s=sum e^(l-m),
// t=sum l*e^(l-m)) + Gumbel-perturbed argmax per row; logits never hit memory.
__global__ void __launch_bounds__(256)
k_logits_partial(const __bf16* __restrict__ hb, const __bf16* __restrict__ woutb,
                 const float* __restrict__ b_out, float* __restrict__ partials, int step) {
  __shared__ __align__(16) char smem[128 * 16 * 6 * 4];  // 48KB: staging (17KB) then reduction
  const int lane = threadIdx.x & 31;
  const int wave = threadIdx.x >> 5;
  const int lh = (lane >> 4) & 1;
  const int r0 = blockIdx.x * 128 + wave * 16;
  const int ntBase = blockIdx.y * TPS_;
  const unsigned ldsBase = (unsigned)(size_t)(void*)smem;  // LDS aperture: low 32 bits = LDS offset

  v16bf afrag[8];
#pragma unroll
  for (int kc = 0; kc < 8; ++kc) afrag[kc] = load_a_frag(hb, H_, r0, kc * 32, lane);

  float m[8], s[8], t[8], pm[8], pv[8];
  int pi[8];
#pragma unroll
  for (int j = 0; j < 8; ++j) { m[j] = -3.0e38f; s[j] = 0.f; t[j] = 0.f; pm[j] = -3.0e38f; pv[j] = 0.f; pi[j] = 0; }

  if (wave == 0)  // prologue: stage first tile into buffer 0
    tdm_load_wout_tile(ldsBase, woutb + (size_t)(ntBase * 16) * H_);

  for (int i = 0; i < TPS_; ++i) {
    const int nt = ntBase + i;
    if (wave == 0) {
      if (i + 1 < TPS_) {
        tdm_load_wout_tile(ldsBase + (unsigned)(((i + 1) & 1) * BUFBYTES_),
                           woutb + (size_t)((nt + 1) * 16) * H_);
        __builtin_amdgcn_s_wait_tensorcnt(1);  // tile i landed; tile i+1 in flight
      } else {
        __builtin_amdgcn_s_wait_tensorcnt(0);
      }
    }
    __syncthreads();  // staged tile i visible to all waves

    const char* buf = smem + (i & 1) * BUFBYTES_;
    v8f acc = {};
#pragma unroll
    for (int kc = 0; kc < 8; ++kc)
      acc = wmma_bf16(afrag[kc], load_b_frag_lds(buf, kc * 32, lane), acc);

    const int col = nt * 16 + (lane & 15);
    const float bias = b_out[col];
#pragma unroll
    for (int j = 0; j < 8; ++j) {
      const int row = r0 + lh * 8 + j;
      float v = acc[j] + bias;
      float p = v + gumbel_noise((unsigned)step, (unsigned)row, (unsigned)col);
      if (p > pm[j]) { pm[j] = p; pv[j] = v; pi[j] = col; }
      if (v > m[j]) {
        float sc = __expf(m[j] - v);
        s[j] = s[j] * sc + 1.0f;
        t[j] = t[j] * sc + v;
        m[j] = v;
      } else {
        float e = __expf(v - m[j]);
        s[j] += e;
        t[j] += e * v;
      }
    }
    __syncthreads();  // everyone done with buffer i before it is restaged
  }

  // intra-block reduction: 128 rows x 16 lane-partials, overlaid on the staging LDS
  float (*red)[16][6] = (float (*)[16][6])smem;
  __syncthreads();
#pragma unroll
  for (int j = 0; j < 8; ++j) {
    int rb = wave * 16 + lh * 8 + j;
    int slot = lane & 15;
    red[rb][slot][0] = m[j];
    red[rb][slot][1] = s[j];
    red[rb][slot][2] = t[j];
    red[rb][slot][3] = pm[j];
    red[rb][slot][4] = pv[j];
    red[rb][slot][5] = __int_as_float(pi[j]);
  }
  __syncthreads();

  if (threadIdx.x < 128) {
    const int rb = threadIdx.x;
    float M = -3.0e38f, S = 0.f, Tt = 0.f, PM = -3.0e38f, PV = 0.f;
    int PI = 0;
    for (int q = 0; q < 16; ++q) {
      float m2 = red[rb][q][0], s2 = red[rb][q][1], t2 = red[rb][q][2];
      float pm2 = red[rb][q][3], pv2 = red[rb][q][4];
      int pi2 = __float_as_int(red[rb][q][5]);
      if (m2 > M) {
        float sc = __expf(M - m2);
        S = S * sc + s2; Tt = Tt * sc + t2; M = m2;
      } else {
        float sc = __expf(m2 - M);
        S += s2 * sc; Tt += t2 * sc;
      }
      if (pm2 > PM) { PM = pm2; PV = pv2; PI = pi2; }
    }
    const size_t row = (size_t)blockIdx.x * 128 + rb;
    float* pp = partials + (row * VSPLIT_ + blockIdx.y) * 8;
    pp[0] = M; pp[1] = S; pp[2] = Tt; pp[3] = PM; pp[4] = PV; pp[5] = __int_as_float(PI);
  }
}

// ---------------- merge V-split partials: outputs + sampled token ----------------
__global__ void k_logits_finalize(const float* __restrict__ partials, float* __restrict__ out,
                                  int* __restrict__ sample, int step) {
  int row = blockIdx.x * blockDim.x + threadIdx.x;
  if (row >= B_) return;
  float M = -3.0e38f, S = 0.f, Tt = 0.f, PM = -3.0e38f, PV = 0.f;
  int PI = 0;
  for (int v = 0; v < VSPLIT_; ++v) {
    const float* pp = partials + ((size_t)row * VSPLIT_ + v) * 8;
    float m2 = pp[0], s2 = pp[1], t2 = pp[2], pm2 = pp[3], pv2 = pp[4];
    int pi2 = __float_as_int(pp[5]);
    if (m2 > M) {
      float sc = __expf(M - m2);
      S = S * sc + s2; Tt = Tt * sc + t2; M = m2;
    } else {
      float sc = __expf(m2 - M);
      S += s2 * sc; Tt += t2 * sc;
    }
    if (pm2 > PM) { PM = pm2; PV = pv2; PI = pi2; }
  }
  float logZ = M + __logf(S);
  sample[row] = PI;
  out[(size_t)row * OC_ + step] = (float)PI;                               // sequence
  out[(size_t)(B_ * OC_) + (size_t)row * OC_ + step] = PV - logZ;          // sampled logprob
  out[(size_t)(2 * B_ * OC_) + (size_t)row * OC_ + step] = logZ - Tt / S;  // entropy
}

// ---------------- launcher ----------------

extern "C" void kernel_launch(void* const* d_in, const int* in_sizes, int n_in,
                              void* d_out, int out_size, void* d_ws, size_t ws_size,
                              hipStream_t stream) {
  const float* x     = (const float*)d_in[0];
  const float* W_in  = (const float*)d_in[1];
  const float* b_in  = (const float*)d_in[2];
  const float* W_out = (const float*)d_in[3];
  const float* b_out = (const float*)d_in[4];
  const float* emb   = (const float*)d_in[5];
  const float* sos   = (const float*)d_in[6];
  const float* W_ih  = (const float*)d_in[7];
  const float* W_hh  = (const float*)d_in[8];
  const float* b_ih  = (const float*)d_in[9];
  const float* b_hh  = (const float*)d_in[10];
  float* out = (float*)d_out;

  char* w = (char*)d_ws;
  auto carve = [&](size_t bytes) -> char* {
    char* p = w;
    w += (bytes + 255) & ~(size_t)255;
    return p;
  };
  __bf16* xb    = (__bf16*)carve((size_t)B_ * F_ * 2);
  __bf16* winb  = (__bf16*)carve((size_t)H_ * F_ * 2);
  __bf16* wihb  = (__bf16*)carve((size_t)4 * H_ * E_ * 2);
  __bf16* whhb  = (__bf16*)carve((size_t)4 * H_ * H_ * 2);
  __bf16* woutb = (__bf16*)carve((size_t)V_ * H_ * 2);
  __bf16* embb  = (__bf16*)carve((size_t)V_ * E_ * 2);
  __bf16* hb    = (__bf16*)carve((size_t)B_ * H_ * 2);
  __bf16* inpb  = (__bf16*)carve((size_t)B_ * E_ * 2);
  float*  c     = (float*)carve((size_t)B_ * H_ * 4);
  float*  gates = (float*)carve((size_t)B_ * 4 * H_ * 4);
  float*  parts = (float*)carve((size_t)B_ * VSPLIT_ * 8 * 4);
  int*    samp  = (int*)carve((size_t)B_ * 4);

  auto cvt = [&](const float* src, __bf16* dst, int n) {
    k_f32_to_bf16<<<(n + 255) / 256, 256, 0, stream>>>(src, dst, n);
  };

  // one-time precision lowering (fp32 -> bf16), zero init of c and d_out pad column
  cvt(x, xb, B_ * F_);
  cvt(W_in, winb, H_ * F_);
  cvt(W_ih, wihb, 4 * H_ * E_);
  cvt(W_hh, whhb, 4 * H_ * H_);
  cvt(W_out, woutb, V_ * H_);
  cvt(emb, embb, V_ * E_);
  k_zero_f32<<<(B_ * H_ + 255) / 256, 256, 0, stream>>>(c, B_ * H_);
  k_zero_f32<<<(3 * B_ * OC_ + 255) / 256, 256, 0, stream>>>(out, 3 * B_ * OC_);

  // h0 = x @ W_in^T + b_in ; inp0 = broadcast(sos)
  k_h0_gemm<<<dim3(B_ / 16, H_ / 64), 128, 0, stream>>>(xb, winb, b_in, hb);
  k_sos_bcast<<<(B_ * E_ + 255) / 256, 256, 0, stream>>>(sos, inpb);

  for (int t = 0; t < T_; ++t) {
    k_gates_gemm<<<dim3(B_ / 16, (4 * H_) / 64), 128, 0, stream>>>(
        inpb, wihb, hb, whhb, b_ih, b_hh, gates);
    k_lstm_pointwise<<<(B_ * H_ + 255) / 256, 256, 0, stream>>>(gates, c, hb);
    k_logits_partial<<<dim3(B_ / 128, VSPLIT_), 256, 0, stream>>>(hb, woutb, b_out, parts, t);
    k_logits_finalize<<<(B_ + 255) / 256, 256, 0, stream>>>(parts, out, samp, t);
    k_embed<<<(B_ * E_ + 255) / 256, 256, 0, stream>>>(embb, samp, inpb);
  }
}
